// LSTMAutoencoder_39436389712617
// MI455X (gfx1250) — matrix operand order repaired and checked by reading, not compiled, not run on hardware
//
#include <hip/hip_runtime.h>
#include <hip/hip_bf16.h>

// ---------------- problem constants ----------------
#define BB   64      // batch
#define TT   256     // time steps
#define II   128     // input dim
#define HH   512     // hidden dim
#define GH   2048    // 4*H gate rows
#define NBLK 16      // blocks in recurrent kernel (each owns 32 h-columns)

typedef __attribute__((ext_vector_type(16))) _Float16 v16h;
typedef __attribute__((ext_vector_type(8)))  _Float16 v8h;
typedef __attribute__((ext_vector_type(8)))  float    v8f;

// ---------------- WMMA helpers (gfx1250, wave32) ----------------
// A fragment: 16x32 fp16, row-major source with leading dim lda (halves).
// Lane L: row = L&15, 16 contiguous K at (L>>4)*16.
__device__ __forceinline__ v16h load_a_frag(const _Float16* a, int lda) {
  const int lane = threadIdx.x & 31;
  const _Float16* p = a + (size_t)(lane & 15) * lda + ((lane >> 4) << 4);
  v8h lo = *(const v8h*)p;
  v8h hi = *(const v8h*)(p + 8);
  v16h r;
#pragma unroll
  for (int i = 0; i < 8; ++i) { r[i] = lo[i]; r[i + 8] = hi[i]; }
  return r;
}

// B fragment: 32x16 (KxN) fp16 from weight matrix stored [N, K] row-major.
// Lane L: column n = L&15 (= weight row), 16 contiguous K at (L>>4)*16.
__device__ __forceinline__ v16h load_b_frag(const _Float16* w, int ldw) {
  const int lane = threadIdx.x & 31;
  return *(const v16h*)(w + (size_t)(lane & 15) * ldw + ((lane >> 4) << 4));
}

__device__ __forceinline__ v8f wmma_f16(v16h a, v16h b, v8f c) {
  return __builtin_amdgcn_wmma_f32_16x16x32_f16(false, a, false, b,
                                                (short)0, c, false, false);
}

__device__ __forceinline__ float sigmoidf_(float x) {
  return 1.0f / (1.0f + __expf(-x));
}

// 2-phase grid barrier over a monotonically increasing counter (zeroed per call
// by hipMemsetAsync before launch; graph-capture safe).
__device__ __forceinline__ void grid_barrier(unsigned* bar, unsigned target) {
  __syncthreads();
  __threadfence();
  if (threadIdx.x == 0) {
    atomicAdd(bar, 1u);
    while (__hip_atomic_load(bar, __ATOMIC_ACQUIRE, __HIP_MEMORY_SCOPE_AGENT) <
           target)
      __builtin_amdgcn_s_sleep(1);
  }
  __syncthreads();
  __threadfence();
}

// ---------------- conversion / packing kernels ----------------
__global__ void cvt_f32_f16_kernel(const float* __restrict__ in,
                                   _Float16* __restrict__ out, int n) {
  int i = blockIdx.x * blockDim.x + threadIdx.x;
  if (i < n) out[i] = (_Float16)in[i];
}

// wcat[r][0..in) = Wih[r][:], wcat[r][in..in+H) = Whh[r][:]  (fp16)
__global__ void pack_wcat_kernel(const float* __restrict__ wih,
                                 const float* __restrict__ whh,
                                 _Float16* __restrict__ wcat, int in_dim) {
  int idx = blockIdx.x * blockDim.x + threadIdx.x;
  int Kc = in_dim + HH;
  int total = GH * Kc;
  if (idx >= total) return;
  int r = idx / Kc, c = idx - r * Kc;
  float v = (c < in_dim) ? wih[(size_t)r * in_dim + c]
                         : whh[(size_t)r * HH + (c - in_dim)];
  wcat[idx] = (_Float16)v;
}

__global__ void bias_sum_kernel(const float* __restrict__ a,
                                const float* __restrict__ b,
                                float* __restrict__ out, int n) {
  int i = blockIdx.x * blockDim.x + threadIdx.x;
  if (i < n) out[i] = a[i] + b[i];
}

// ---------------- recurrent LSTM layer (one launch per layer) ----------------
// grid = NBLK blocks x 256 threads (8 wave32). Block j owns h cols [32j,32j+32).
// Gate GEMM per step: [64, Kc] x [Kc, 2048] with Kc = in_dim + H, A = [x_t | h].
__global__ void lstm_layer_kernel(const _Float16* __restrict__ xsrc,
                                  int in_dim, int x_is_seq,
                                  const _Float16* __restrict__ wcat, // [GH, Kc]
                                  const float* __restrict__ bias,    // [GH]
                                  _Float16* __restrict__ hbuf,       // [B, H]
                                  _Float16* __restrict__ yseq,       // [B,T,H] or null
                                  unsigned* __restrict__ bar) {
  extern __shared__ char smem[];
  const int Kc = in_dim + HH;
  _Float16* xcat = (_Float16*)smem;                                  // [64][Kc]
  float* gbuf = (float*)(smem + (size_t)64 * Kc * sizeof(_Float16)); // [4][64][33]
  float* cbuf = gbuf + 4 * 64 * 33;                                  // [64][32]

  const int tid = threadIdx.x;
  const int lane = tid & 31;
  const int wv = tid >> 5;            // wave 0..7
  const int j0 = blockIdx.x * 32;     // h-column base of this block
  const int grp = wv >> 1;            // gate group (i,f,g,o)
  const int nc0 = (wv & 1) * 16;      // local 16-col slab within the 32 cols
  const int wrow0 = grp * HH + j0 + nc0;  // weight-row / gate-row base

  // init cell state + our slice of the shared h buffer
  for (int i = tid; i < 64 * 32; i += 256) cbuf[i] = 0.0f;
  for (int i = tid; i < 64 * 32; i += 256) {
    int m = i >> 5, c = i & 31;
    hbuf[(size_t)m * HH + j0 + c] = (_Float16)0.0f;
  }
  unsigned nbar = 0;
  grid_barrier(bar, ++nbar * gridDim.x);

  const float bval = bias[wrow0 + (lane & 15)];
  const int xch = in_dim >> 3;   // 8-half chunks per x row
  const int hch = HH >> 3;

  for (int t = 0; t < TT; ++t) {
    // ---- stage A = [x_t | h_{t-1}] into LDS (fp16) ----
    for (int ci = tid; ci < 64 * xch; ci += 256) {
      int m = ci / xch, co = (ci - m * xch) << 3;
      const _Float16* src = x_is_seq
          ? (xsrc + ((size_t)m * TT + t) * in_dim + co)
          : (xsrc + (size_t)m * in_dim + co);
      *(v8h*)(xcat + (size_t)m * Kc + co) = *(const v8h*)src;
    }
    for (int ci = tid; ci < 64 * hch; ci += 256) {
      int m = ci / hch, co = (ci - m * hch) << 3;
      *(v8h*)(xcat + (size_t)m * Kc + in_dim + co) =
          *(const v8h*)(hbuf + (size_t)m * HH + co);
    }
    // everyone must finish reading h(t-1) before anyone overwrites it
    grid_barrier(bar, ++nbar * gridDim.x);

    // ---- gate GEMM: this wave -> 64 rows x 16 gate-cols ----
    v8f acc[4];
#pragma unroll
    for (int mt = 0; mt < 4; ++mt)
#pragma unroll
      for (int i = 0; i < 8; ++i) acc[mt][i] = bval;

    const _Float16* wbase = wcat + (size_t)wrow0 * Kc;
#pragma unroll 2
    for (int k0 = 0; k0 < Kc; k0 += 32) {
      // speculative prefetch of the next weight slab (harmless past the end)
      __builtin_prefetch(wbase + (size_t)(lane & 15) * Kc + k0 + 32, 0, 1);
      v16h bf = load_b_frag(wbase + k0, Kc);
#pragma unroll
      for (int mt = 0; mt < 4; ++mt) {
        v16h af = load_a_frag(xcat + (size_t)mt * 16 * Kc + k0, Kc);
        acc[mt] = wmma_f16(af, bf, acc[mt]);
      }
    }

    // ---- spill gates to LDS (padded stride 33 to dodge bank conflicts) ----
#pragma unroll
    for (int mt = 0; mt < 4; ++mt)
#pragma unroll
      for (int v = 0; v < 8; ++v) {
        int m = mt * 16 + v + ((lane >> 4) << 3);
        int cl = nc0 + (lane & 15);
        gbuf[((size_t)grp * 64 + m) * 33 + cl] = acc[mt][v];
      }
    __syncthreads();

    // ---- elementwise LSTM update for our 64x32 slice ----
    {
      int jl = tid & 31;
      int m0 = (tid >> 5) * 8;
      if (yseq) {
#pragma unroll
        for (int mm = 0; mm < 8; ++mm) {
          int m = m0 + mm;
          float ig = sigmoidf_(gbuf[((size_t)0 * 64 + m) * 33 + jl]);
          float fg = sigmoidf_(gbuf[((size_t)1 * 64 + m) * 33 + jl]);
          float gg = tanhf(gbuf[((size_t)2 * 64 + m) * 33 + jl]);
          float og = sigmoidf_(gbuf[((size_t)3 * 64 + m) * 33 + jl]);
          float c = fg * cbuf[m * 32 + jl] + ig * gg;
          cbuf[m * 32 + jl] = c;
          float h = og * tanhf(c);
          _Float16 hh = (_Float16)h;
          hbuf[(size_t)m * HH + j0 + jl] = hh;
          yseq[((size_t)m * TT + t) * HH + j0 + jl] = hh;
        }
      } else {
#pragma unroll
        for (int mm = 0; mm < 8; ++mm) {
          int m = m0 + mm;
          float ig = sigmoidf_(gbuf[((size_t)0 * 64 + m) * 33 + jl]);
          float fg = sigmoidf_(gbuf[((size_t)1 * 64 + m) * 33 + jl]);
          float gg = tanhf(gbuf[((size_t)2 * 64 + m) * 33 + jl]);
          float og = sigmoidf_(gbuf[((size_t)3 * 64 + m) * 33 + jl]);
          float c = fg * cbuf[m * 32 + jl] + ig * gg;
          cbuf[m * 32 + jl] = c;
          float h = og * tanhf(c);
          hbuf[(size_t)m * HH + j0 + jl] = (_Float16)h;
        }
      }
    }
    grid_barrier(bar, ++nbar * gridDim.x);  // publish h(t)
  }
}

// ---------------- reconstruction head: [M,512]f16 x [128,512]^T + b ----------
__global__ void recon_head_kernel(const _Float16* __restrict__ A,
                                  const _Float16* __restrict__ W,
                                  const float* __restrict__ bias,
                                  float* __restrict__ out) {
  const int tid = threadIdx.x, lane = tid & 31, wv = tid >> 5;
  const int m0 = blockIdx.x * 64;
  const int n0 = wv * 16;
  const float bval = bias[n0 + (lane & 15)];
  v8f acc[4];
#pragma unroll
  for (int mt = 0; mt < 4; ++mt)
#pragma unroll
    for (int i = 0; i < 8; ++i) acc[mt][i] = bval;

  const _Float16* wbase = W + (size_t)n0 * HH;
  const _Float16* abase = A + (size_t)m0 * HH;
#pragma unroll 2
  for (int k0 = 0; k0 < HH; k0 += 32) {
    v16h bf = load_b_frag(wbase + k0, HH);
#pragma unroll
    for (int mt = 0; mt < 4; ++mt) {
      v16h af = load_a_frag(abase + (size_t)mt * 16 * HH + k0, HH);
      acc[mt] = wmma_f16(af, bf, acc[mt]);
    }
  }
#pragma unroll
  for (int mt = 0; mt < 4; ++mt)
#pragma unroll
    for (int v = 0; v < 8; ++v) {
      int m = m0 + mt * 16 + v + ((lane >> 4) << 3);
      int n = n0 + (lane & 15);
      out[(size_t)m * II + n] = acc[mt][v];
    }
}

// ---------------- prediction head: softmax(tanh(latent @ Wp^T + bp)) ---------
__global__ void pred_head_kernel(const _Float16* __restrict__ latent, // [64,512]
                                 const float* __restrict__ W,         // [10,512]
                                 const float* __restrict__ bias,      // [10]
                                 float* __restrict__ out) {           // [64,10]
  __shared__ float logits[BB][12];
  int tid = threadIdx.x;
  for (int p = tid; p < BB * 10; p += blockDim.x) {
    int b = p / 10, c = p - b * 10;
    float acc = bias[c];
    for (int k = 0; k < HH; ++k)
      acc += (float)latent[(size_t)b * HH + k] * W[(size_t)c * HH + k];
    logits[b][c] = tanhf(acc);
  }
  __syncthreads();
  for (int b = tid; b < BB; b += blockDim.x) {
    float mx = -1e30f;
    for (int c = 0; c < 10; ++c) mx = fmaxf(mx, logits[b][c]);
    float e[10], s = 0.f;
    for (int c = 0; c < 10; ++c) { e[c] = __expf(logits[b][c] - mx); s += e[c]; }
    float inv = 1.0f / s;
    for (int c = 0; c < 10; ++c) out[b * 10 + c] = e[c] * inv;
  }
}

// ---------------- host launcher ----------------
extern "C" void kernel_launch(void* const* d_in, const int* in_sizes, int n_in,
                              void* d_out, int out_size, void* d_ws, size_t ws_size,
                              hipStream_t stream) {
  const float* x        = (const float*)d_in[0];
  const float* enc_Wih0 = (const float*)d_in[1];
  const float* enc_Wih1 = (const float*)d_in[2];
  const float* enc_Whh  = (const float*)d_in[3];
  const float* enc_bih  = (const float*)d_in[4];
  const float* enc_bhh  = (const float*)d_in[5];
  const float* dec_Wih  = (const float*)d_in[6];
  const float* dec_Whh  = (const float*)d_in[7];
  const float* dec_bih  = (const float*)d_in[8];
  const float* dec_bhh  = (const float*)d_in[9];
  const float* lin_W    = (const float*)d_in[10];
  const float* lin_b    = (const float*)d_in[11];
  const float* pred_W   = (const float*)d_in[12];
  const float* pred_b   = (const float*)d_in[13];

  char* ws = (char*)d_ws;
  size_t off = 0;
  auto carve = [&](size_t bytes) -> char* {
    char* p = ws + off;
    off = (off + bytes + 255) & ~(size_t)255;
    return p;
  };
  unsigned* cnt      = (unsigned*)carve(64);                       // 4 barrier counters
  _Float16* x16      = (_Float16*)carve((size_t)BB * TT * II * 2);
  _Float16* bufA     = (_Float16*)carve((size_t)BB * TT * HH * 2); // enc0 out / dec0 out
  _Float16* bufB     = (_Float16*)carve((size_t)BB * TT * HH * 2); // dec1 out
  _Float16* hbuf     = (_Float16*)carve((size_t)BB * HH * 2);
  _Float16* latent   = (_Float16*)carve((size_t)BB * HH * 2);
  _Float16* wcat0    = (_Float16*)carve((size_t)GH * (II + HH) * 2);
  _Float16* wcat1    = (_Float16*)carve((size_t)GH * (HH + HH) * 2);
  _Float16* wcat2    = (_Float16*)carve((size_t)GH * (HH + HH) * 2);
  _Float16* wcat3    = (_Float16*)carve((size_t)GH * (HH + HH) * 2);
  _Float16* linW16   = (_Float16*)carve((size_t)II * HH * 2);
  float*    bias0    = (float*)carve((size_t)GH * 4);
  float*    bias1    = (float*)carve((size_t)GH * 4);
  float*    bias2    = (float*)carve((size_t)GH * 4);
  float*    bias3    = (float*)carve((size_t)GH * 4);

  hipMemsetAsync(cnt, 0, 64, stream);

  // fp32 -> fp16 conversions
  {
    int n = BB * TT * II;
    cvt_f32_f16_kernel<<<(n + 255) / 256, 256, 0, stream>>>(x, x16, n);
    n = II * HH;
    cvt_f32_f16_kernel<<<(n + 255) / 256, 256, 0, stream>>>(lin_W, linW16, n);
  }
  // pack [Wih | Whh] per layer
  {
    int n0 = GH * (II + HH);
    pack_wcat_kernel<<<(n0 + 255) / 256, 256, 0, stream>>>(enc_Wih0, enc_Whh, wcat0, II);
    int n1 = GH * (HH + HH);
    pack_wcat_kernel<<<(n1 + 255) / 256, 256, 0, stream>>>(
        enc_Wih1, enc_Whh + (size_t)GH * HH, wcat1, HH);
    pack_wcat_kernel<<<(n1 + 255) / 256, 256, 0, stream>>>(
        dec_Wih, dec_Whh, wcat2, HH);
    pack_wcat_kernel<<<(n1 + 255) / 256, 256, 0, stream>>>(
        dec_Wih + (size_t)GH * HH, dec_Whh + (size_t)GH * HH, wcat3, HH);
  }
  // fused biases
  bias_sum_kernel<<<(GH + 255) / 256, 256, 0, stream>>>(enc_bih, enc_bhh, bias0, GH);
  bias_sum_kernel<<<(GH + 255) / 256, 256, 0, stream>>>(enc_bih + GH, enc_bhh + GH, bias1, GH);
  bias_sum_kernel<<<(GH + 255) / 256, 256, 0, stream>>>(dec_bih, dec_bhh, bias2, GH);
  bias_sum_kernel<<<(GH + 255) / 256, 256, 0, stream>>>(dec_bih + GH, dec_bhh + GH, bias3, GH);

  // dynamic LDS: xcat + gate spill + cell state
  auto smem_bytes = [](int in_dim) -> size_t {
    size_t Kc = (size_t)in_dim + HH;
    return 64 * Kc * 2 + 4 * 64 * 33 * 4 + 64 * 32 * 4;
  };
  (void)hipFuncSetAttribute((const void*)lstm_layer_kernel,
                            hipFuncAttributeMaxDynamicSharedMemorySize,
                            (int)smem_bytes(HH));

  // encoder layer 0: x -> bufA
  lstm_layer_kernel<<<NBLK, 256, smem_bytes(II), stream>>>(
      x16, II, 1, wcat0, bias0, hbuf, bufA, cnt + 0);
  // encoder layer 1: bufA -> (latent only)
  lstm_layer_kernel<<<NBLK, 256, smem_bytes(HH), stream>>>(
      bufA, HH, 1, wcat1, bias1, hbuf, (_Float16*)nullptr, cnt + 1);
  // latent = final h of enc layer 1
  hipMemcpyAsync(latent, hbuf, (size_t)BB * HH * 2, hipMemcpyDeviceToDevice, stream);
  // decoder layer 0: latent (constant over t) -> bufA
  lstm_layer_kernel<<<NBLK, 256, smem_bytes(HH), stream>>>(
      latent, HH, 0, wcat2, bias2, hbuf, bufA, cnt + 2);
  // decoder layer 1: bufA -> bufB
  lstm_layer_kernel<<<NBLK, 256, smem_bytes(HH), stream>>>(
      bufA, HH, 1, wcat3, bias3, hbuf, bufB, cnt + 3);

  // heads
  float* rec_out = (float*)d_out;                   // [B,T,I]
  float* pred_out = rec_out + (size_t)BB * TT * II; // [B,C]
  recon_head_kernel<<<(BB * TT) / 64, 256, 0, stream>>>(bufB, linW16, lin_b, rec_out);
  pred_head_kernel<<<1, 256, 0, stream>>>(latent, pred_W, pred_b, pred_out);
}
